// GCNModel_66726611910685
// MI455X (gfx1250) — matrix operand order repaired and checked by reading, not compiled, not run on hardware
//
#include <hip/hip_runtime.h>

#define Nn 100000
#define Ff 128
#define Hh 128
#define Rr 4
#define Bb 2
#define Cc 3
#define Ee 1600000

typedef __bf16 bf16_t;
typedef __attribute__((ext_vector_type(16))) __bf16 v16bf;
typedef __attribute__((ext_vector_type(8)))  __bf16 v8bf;
typedef __attribute__((ext_vector_type(8)))  float  v8f;
typedef __attribute__((ext_vector_type(4)))  float  v4f;

// V_r = sum_b comp[r,b] * W[b]  -> stored transposed (out-major) in bf16:
// Vt[r][o][i] = sum_b comp[r*B+b] * W[((b*F)+i)*H + o]
__global__ __launch_bounds__(256) void basis_kernel(const float* __restrict__ W,
                                                    const float* __restrict__ comp,
                                                    bf16_t* __restrict__ Vt) {
  int idx = blockIdx.x * 256 + threadIdx.x;
  if (idx >= Rr * Hh * Ff) return;
  int i = idx % Ff;
  int o = (idx / Ff) % Hh;
  int r = idx / (Ff * Hh);
  float acc = 0.f;
#pragma unroll
  for (int b = 0; b < Bb; ++b)
    acc += comp[r * Bb + b] * W[(b * Ff + i) * Hh + o];
  Vt[idx] = (bf16_t)acc;
}

__global__ __launch_bounds__(256) void cvt_kernel(const float* __restrict__ X,
                                                  bf16_t* __restrict__ Xb, int n) {
  int idx = blockIdx.x * 256 + threadIdx.x;
  if (idx < n) Xb[idx] = (bf16_t)X[idx];
}

// S[M=100000, N=128] = A[M,128](bf16) * Bt[N,128](bf16, transposed), f32 accum.
// One wave32 per 16x16 output tile; K=128 -> 4 x v_wmma_f32_16x16x32_bf16.
__global__ __launch_bounds__(256) void gemm_kernel(const bf16_t* __restrict__ A,
                                                   const bf16_t* __restrict__ Bt,
                                                   float* __restrict__ S) {
  const int lane = threadIdx.x & 31;
  const int wave = (blockIdx.x * 256 + threadIdx.x) >> 5;
  const int tm = wave >> 3;          // 0..6249
  const int tn = wave & 7;           // 0..7
  const int hl = lane >> 4;          // half-wave select
  const int ll = lane & 15;
  // A fragment: lane holds row (ll); k-run of 16 at hl*16
  const bf16_t* arow = A  + (size_t)(tm * 16 + ll) * Ff;
  // B fragment: lane holds column (ll); k-run of 16 at hl*16
  const bf16_t* brow = Bt + (size_t)(tn * 16 + ll) * Ff;
  const int aoff = hl * 16;
  const int boff = hl * 16;
  v8f c = {};
#pragma unroll
  for (int kb = 0; kb < Ff; kb += 32) {
    union { v16bf v; v8bf h[2]; } af;
    af.h[0] = *(const v8bf*)(arow + kb + aoff);
    af.h[1] = *(const v8bf*)(arow + kb + aoff + 8);
    v16bf bvec = *(const v16bf*)(brow + kb + boff);
    c = __builtin_amdgcn_wmma_f32_16x16x32_bf16(
        /*neg_a=*/false, af.v, /*neg_b=*/false, bvec,
        /*c_mod=*/(short)0, c, /*reuse_a=*/false, /*reuse_b=*/false);
  }
  // D layout: VGPR v -> row tm*16 + v + 8*hl, col tn*16 + ll
  float* outp = S + (size_t)(tm * 16 + hl * 8) * Hh + tn * 16 + ll;
#pragma unroll
  for (int v = 0; v < 8; ++v) outp[v * Hh] = c[v];
}

__device__ __forceinline__ void gatomic_add(float* p, float v) {
  asm volatile("global_atomic_add_f32 %0, %1, off scope:SCOPE_DEV"
               :: "v"(p), "v"(v) : "memory");
}

// One wave per edge: acc[dst] += ew * S[src]  (128 floats -> float4 per lane)
__global__ __launch_bounds__(256) void edge_kernel(const float* __restrict__ S,
                                                   const int* __restrict__ src,
                                                   const int* __restrict__ dst,
                                                   const float* __restrict__ ew,
                                                   float* __restrict__ acc) {
  const int lane = threadIdx.x & 31;
  int e = (int)((blockIdx.x * 256u + threadIdx.x) >> 5);
  if (e >= Ee) return;
  e = __builtin_amdgcn_readfirstlane(e);   // force scalar loads of edge data
  const int   s = src[e];
  const int   d = dst[e];
  const float w = ew[e];
  const v4f val = *((const v4f*)(S + (size_t)s * Hh) + lane);
  float* dp = acc + (size_t)d * Hh + lane * 4;
  gatomic_add(dp + 0, w * val.x);
  gatomic_add(dp + 1, w * val.y);
  gatomic_add(dp + 2, w * val.z);
  gatomic_add(dp + 3, w * val.w);
}

__global__ __launch_bounds__(256) void tanh_kernel(const float* __restrict__ a,
                                                   float* __restrict__ h, int n) {
  int idx = blockIdx.x * 256 + threadIdx.x;
  if (idx < n) h[idx] = tanhf(a[idx]);
}

// out[n, :3] = tanh(acc[n,:]) @ Wc + bc   (tanh fused here for layer 2)
__global__ __launch_bounds__(256) void clf_kernel(const float* __restrict__ acc,
                                                  const float* __restrict__ Wc,
                                                  const float* __restrict__ bc,
                                                  float* __restrict__ out) {
  int n = blockIdx.x * 256 + threadIdx.x;
  if (n >= Nn) return;
  float a0 = bc[0], a1 = bc[1], a2 = bc[2];
  const float* row = acc + (size_t)n * Hh;
#pragma unroll 4
  for (int k = 0; k < Hh; ++k) {
    float x = tanhf(row[k]);
    a0 += x * Wc[k * Cc + 0];
    a1 += x * Wc[k * Cc + 1];
    a2 += x * Wc[k * Cc + 2];
  }
  out[n * Cc + 0] = a0;
  out[n * Cc + 1] = a1;
  out[n * Cc + 2] = a2;
}

extern "C" void kernel_launch(void* const* d_in, const int* in_sizes, int n_in,
                              void* d_out, int out_size, void* d_ws, size_t ws_size,
                              hipStream_t stream) {
  (void)in_sizes; (void)n_in; (void)out_size; (void)ws_size;
  const float* emb = (const float*)d_in[0];
  const float* W1  = (const float*)d_in[1];
  const float* W1c = (const float*)d_in[2];
  const float* W2  = (const float*)d_in[3];
  const float* W2c = (const float*)d_in[4];
  const float* Wc  = (const float*)d_in[5];
  const float* bcv = (const float*)d_in[6];
  const float* ew  = (const float*)d_in[7];
  const int*   src = (const int*)d_in[8];
  const int*   dst = (const int*)d_in[9];
  float* out = (float*)d_out;

  // workspace layout (all 256B aligned by construction)
  char* base = (char*)d_ws;
  const size_t VT_B  = (size_t)Rr * Hh * Ff * 2;   //   131072
  const size_t XB_B  = (size_t)Nn * Ff * 2;        // 25600000
  const size_t F32_B = (size_t)Nn * Hh * 4;        // 51200000
  bf16_t* Vt  = (bf16_t*)(base);
  bf16_t* Xb  = (bf16_t*)(base + VT_B);
  float*  S   = (float*)(base + VT_B + XB_B);
  float*  acc = (float*)(base + VT_B + XB_B + F32_B);
  float*  hb  = (float*)(base + VT_B + XB_B + 2 * F32_B);

  auto layer = [&](const float* Xsrc, const float* Wb, const float* Wcp) {
    basis_kernel<<<(Rr * Hh * Ff + 255) / 256, 256, 0, stream>>>(Wb, Wcp, Vt);
    cvt_kernel<<<(Nn * Ff + 255) / 256, 256, 0, stream>>>(Xsrc, Xb, Nn * Ff);
    hipMemsetAsync(acc, 0, F32_B, stream);
    for (int r = 0; r < Rr; ++r) {
      gemm_kernel<<<(Nn / 16) * 8 / 8, 256, 0, stream>>>(Xb, Vt + (size_t)r * Hh * Ff, S);
      edge_kernel<<<(Ee * 32) / 256, 256, 0, stream>>>(S, src + (size_t)r * Ee,
                                                       dst + (size_t)r * Ee,
                                                       ew + (size_t)r * Ee, acc);
    }
  };

  layer(emb, W1, W1c);                                         // gc1 (pre-act in acc)
  tanh_kernel<<<(Nn * Hh + 255) / 256, 256, 0, stream>>>(acc, hb, Nn * Hh);
  layer(hb, W2, W2c);                                          // gc2 (pre-act in acc)
  clf_kernel<<<(Nn + 255) / 256, 256, 0, stream>>>(acc, Wc, bcv, out);
}
